// MEGConformerLayer_29609504539067
// MI455X (gfx1250) — compile-verified, matrix operand order checked by reading, not run on hardware
//
#include <hip/hip_runtime.h>
#include <hip/hip_bf16.h>
#include <math.h>

// Problem constants (match reference)
#define B_    8
#define S_    1024
#define D_    512
#define H_    8
#define DH_   64
#define FF_   1024
#define SPAN_ 32

typedef __attribute__((ext_vector_type(8)))  _Float16 v8h;
typedef __attribute__((ext_vector_type(16))) _Float16 v16h;
typedef __attribute__((ext_vector_type(8)))  float    v8f;

// Async global->LDS DMA path (gfx1250). The load itself is emitted via inline
// asm (VFLAT: vdst = LDS-address VGPR, vaddr = 64-bit global address, no
// saddr -> "off"); completion is ordered with the probe-confirmed
// s_wait_asynccnt builtin. Device-compile only; host pass and toolchains
// without the builtin use the register-staged fallback pipeline.
#if defined(__HIP_DEVICE_COMPILE__) && \
    __has_builtin(__builtin_amdgcn_s_wait_asynccnt) && \
    __has_builtin(__builtin_amdgcn_global_load_async_to_lds_b128)
#define ASYNC_LDS 1
#else
#define ASYNC_LDS 0
#endif

// ---------------------------------------------------------------------------
// WMMA GEMM: C[M,N] = act(scale * A[M,K] @ Bt[N,K]^T + bias) + Res
// A, Bt f16; C f32 or f16. Batched over blockIdx.z = zb*nh + zh with
// independent strides so per-head submatrices (lda=D, col offset h*64) work.
// Block tile 128x128, 8 waves (wave32) in 2x4; each wave 64x32 = 4x2 WMMA
// tiles. Double-buffered LDS, one barrier per K step, prefetch of tile k+2.
// Staging uses GLOBAL_LOAD_ASYNC_TO_LDS_B128 (ASYNCcnt) when available.
// Requirements: K % 32 == 0; out-of-range staging rows are clamped (their
// results are masked at store time).
// ---------------------------------------------------------------------------
#define BM 128
#define BN 128
#define BK 32
#define LDSS 48   // f16 row stride in LDS: 96B, 16B-aligned, conflict-skewed

#if ASYNC_LDS
__device__ __forceinline__ void stage_async_b128(const _Float16* g, _Float16* l) {
  // Low 32 bits of a generic LDS pointer are the LDS offset (flat->LDS
  // aperture truncation, ISA 10.2).
  const unsigned lds_off = (unsigned)(__SIZE_TYPE__)l;
  const unsigned long long gaddr = (unsigned long long)(__SIZE_TYPE__)g;
  asm volatile("global_load_async_to_lds_b128 %0, %1, off"
               :
               : "v"(lds_off), "v"(gaddr)
               : "memory");
}
#endif

__device__ __forceinline__ v16h frag_ld(const _Float16* row, int lane) {
  // 16-bit A/B operand layout: lanes 0-15 hold K=0..7 (vgpr0-3) + K=16..23
  // (vgpr4-7); lanes 16-31 hold K=8..15 + K=24..31 for the same row.
  const _Float16* p = row + ((lane >> 4) << 3);
  v8h lo = *(const v8h*)(p);
  v8h hi = *(const v8h*)(p + 16);
  return __builtin_shufflevector(lo, hi, 0,1,2,3,4,5,6,7,8,9,10,11,12,13,14,15);
}

__device__ __forceinline__ void wave_mma(
    const _Float16 (*__restrict__ As)[LDSS],
    const _Float16 (*__restrict__ Bs)[LDSS],
    int wm, int wn, int lane, v8f (&acc)[4][2])
{
  const int l15 = lane & 15;
  v16h af[4], bf[2];
#pragma unroll
  for (int t = 0; t < 4; ++t)
    af[t] = frag_ld(As[wm * 64 + t * 16 + l15], lane);
#pragma unroll
  for (int t = 0; t < 2; ++t)
    bf[t] = frag_ld(Bs[wn * 32 + t * 16 + l15], lane);
#pragma unroll
  for (int tm = 0; tm < 4; ++tm)
#pragma unroll
    for (int tn = 0; tn < 2; ++tn)
      acc[tm][tn] = __builtin_amdgcn_wmma_f32_16x16x32_f16(
          false, af[tm], false, bf[tn], (short)0, acc[tm][tn], false, false);
}

template<bool BIAS, bool RES, bool SILU, bool OUTF16>
__global__ __launch_bounds__(256) void gemm_wmma_kernel(
    const _Float16* __restrict__ A,  int lda, long sAb, long sAh,
    const _Float16* __restrict__ Bt, int ldb, long sBb, long sBh,
    void* __restrict__ Cv,           int ldc, long sCb, long sCh,
    const float* __restrict__ Res,   int ldr, long sRb, long sRh,
    const float* __restrict__ bias,
    int M, int N, int K, int nh, float scale)
{
  __shared__ __align__(16) _Float16 As[2][BM][LDSS];
  __shared__ __align__(16) _Float16 Bs[2][BN][LDSS];

  const int z  = blockIdx.z;
  const int zb = z / nh, zh = z % nh;
  A  += (long)zb * sAb + (long)zh * sAh;
  Bt += (long)zb * sBb + (long)zh * sBh;
  const long coff = (long)zb * sCb + (long)zh * sCh;
  const long roff = (long)zb * sRb + (long)zh * sRh;

  const int tid  = threadIdx.x;
  const int lane = tid & 31;
  const int wave = tid >> 5;
  const int wm   = wave >> 2;   // 0..1 -> 64-row band
  const int wn   = wave & 3;    // 0..3 -> 32-col band
  const int bm   = blockIdx.y * BM;
  const int bn   = blockIdx.x * BN;

  // Staging: each thread owns one row-half (16 f16 = 2 x b128) of A and B.
  const int r = tid >> 1;
  const int c = (tid & 1) << 4;
  const int ra = bm + r < M ? bm + r : M - 1;   // clamp; masked at epilogue
  const int rb = bn + r < N ? bn + r : N - 1;
  const _Float16* pa = A  + (long)ra * lda + c;
  const _Float16* pb = Bt + (long)rb * ldb + c;

  v8f acc[4][2] = {};
  int buf = 0;
  const int nk = K / BK;

#if ASYNC_LDS
  // ---- async DMA pipeline: global -> LDS with no VGPR staging ----
  stage_async_b128(pa,     &As[0][r][c]);
  stage_async_b128(pa + 8, &As[0][r][c + 8]);
  stage_async_b128(pb,     &Bs[0][r][c]);
  stage_async_b128(pb + 8, &Bs[0][r][c + 8]);
  __builtin_amdgcn_s_wait_asynccnt(0);
  __syncthreads();

  for (int it = 1; it < nk; ++it) {
    const int k0 = it * BK;
    const int nb = buf ^ 1;
    // Next tile's copies run underneath the WMMA burst. Buffer nb was last
    // read before the previous barrier, so the DMA writes are race-free.
    stage_async_b128(pa + k0,     &As[nb][r][c]);
    stage_async_b128(pa + k0 + 8, &As[nb][r][c + 8]);
    stage_async_b128(pb + k0,     &Bs[nb][r][c]);
    stage_async_b128(pb + k0 + 8, &Bs[nb][r][c + 8]);
    __builtin_prefetch(pa + k0 + BK, 0, 1);
    __builtin_prefetch(pb + k0 + BK, 0, 1);

    wave_mma(As[buf], Bs[buf], wm, wn, lane, acc);

    __builtin_amdgcn_s_wait_asynccnt(0);
    __syncthreads();
    buf = nb;
  }
#else
  // ---- register-staged pipeline (fallback) ----
  v8h a0 = *(const v8h*)pa,        a1 = *(const v8h*)(pa + 8);
  v8h b0 = *(const v8h*)pb,        b1 = *(const v8h*)(pb + 8);
  *(v8h*)&As[0][r][c] = a0;  *(v8h*)&As[0][r][c + 8] = a1;
  *(v8h*)&Bs[0][r][c] = b0;  *(v8h*)&Bs[0][r][c + 8] = b1;
  __syncthreads();

  for (int it = 1; it < nk; ++it) {
    const int k0 = it * BK;
    a0 = *(const v8h*)(pa + k0);  a1 = *(const v8h*)(pa + k0 + 8);
    b0 = *(const v8h*)(pb + k0);  b1 = *(const v8h*)(pb + k0 + 8);
    __builtin_prefetch(pa + k0 + BK, 0, 1);
    __builtin_prefetch(pb + k0 + BK, 0, 1);

    wave_mma(As[buf], Bs[buf], wm, wn, lane, acc);

    const int nb = buf ^ 1;
    *(v8h*)&As[nb][r][c] = a0;  *(v8h*)&As[nb][r][c + 8] = a1;
    *(v8h*)&Bs[nb][r][c] = b0;  *(v8h*)&Bs[nb][r][c + 8] = b1;
    __syncthreads();
    buf = nb;
  }
#endif
  wave_mma(As[buf], Bs[buf], wm, wn, lane, acc);

  // Epilogue. C layout: lanes 0-15 -> M=r, N=lane; lanes 16-31 -> M=r+8.
  float*    C32 = (float*)Cv;
  _Float16* C16 = (_Float16*)Cv;
  const int nlo = lane & 15;
  const int mhi = (lane >> 4) << 3;
#pragma unroll
  for (int tm = 0; tm < 4; ++tm) {
#pragma unroll
    for (int tn = 0; tn < 2; ++tn) {
      const int n = bn + wn * 32 + tn * 16 + nlo;
      if (n >= N) continue;
      const float bval = BIAS ? bias[n] : 0.f;
#pragma unroll
      for (int rr = 0; rr < 8; ++rr) {
        const int m = bm + wm * 64 + tm * 16 + mhi + rr;
        if (m >= M) continue;
        float v = acc[tm][tn][rr] * scale;
        if (BIAS) v += bval;
        if (SILU) v = v / (1.f + __expf(-v));
        if (RES)  v += Res[roff + (long)m * ldr + n];
        const long ci = coff + (long)m * ldc + n;
        if (OUTF16) C16[ci] = (_Float16)v;
        else        C32[ci] = v;
      }
    }
  }
}

// ---------------------------------------------------------------------------
// LayerNorm over D=512, one block (8 waves) per row, f16 output (GEMM operand)
// ---------------------------------------------------------------------------
__global__ __launch_bounds__(256) void ln_f16_kernel(
    const float* __restrict__ x, const float* __restrict__ g,
    const float* __restrict__ b, _Float16* __restrict__ out)
{
  __shared__ float red[8];
  const long row = blockIdx.x;
  const float* xr = x + row * D_;
  const int tid = threadIdx.x;
  const float v0 = xr[tid], v1 = xr[tid + 256];

  float s = v0 + v1;
  for (int o = 16; o > 0; o >>= 1) s += __shfl_down(s, o);
  if ((tid & 31) == 0) red[tid >> 5] = s;
  __syncthreads();
  if (tid == 0) { float t = 0; for (int i = 0; i < 8; ++i) t += red[i]; red[0] = t; }
  __syncthreads();
  const float mean = red[0] * (1.f / D_);
  const float d0 = v0 - mean, d1 = v1 - mean;

  float vs = d0 * d0 + d1 * d1;
  __syncthreads();
  for (int o = 16; o > 0; o >>= 1) vs += __shfl_down(vs, o);
  if ((tid & 31) == 0) red[tid >> 5] = vs;
  __syncthreads();
  if (tid == 0) { float t = 0; for (int i = 0; i < 8; ++i) t += red[i]; red[0] = t; }
  __syncthreads();
  const float inv = rsqrtf(red[0] * (1.f / D_) + 1e-5f);

  out[row * D_ + tid]       = (_Float16)(d0 * inv * g[tid] + b[tid]);
  out[row * D_ + tid + 256] = (_Float16)(d1 * inv * g[tid + 256] + b[tid + 256]);
}

// Depthwise conv (K=3, same-pad) + bias + SiLU, f16 in/out
__global__ void dwconv_silu_kernel(const _Float16* __restrict__ xn,
                                   const float* __restrict__ w,
                                   const float* __restrict__ bias,
                                   _Float16* __restrict__ out, long total)
{
  const long i = (long)blockIdx.x * blockDim.x + threadIdx.x;
  if (i >= total) return;
  const int  d  = (int)(i % D_);
  const long bs = i / D_;
  const int  s  = (int)(bs % S_);
  const long b  = bs / S_;
  float acc = bias[d];
#pragma unroll
  for (int t = 0; t < 3; ++t) {
    const int ss = s + t - 1;
    if (ss >= 0 && ss < S_)
      acc += (float)xn[(b * S_ + ss) * D_ + d] * w[d * 3 + t];
  }
  acc = acc / (1.f + __expf(-acc));
  out[i] = (_Float16)acc;
}

__global__ void f32_to_f16_kernel(const float* __restrict__ in,
                                  _Float16* __restrict__ out, long n)
{
  const long i = (long)blockIdx.x * blockDim.x + threadIdx.x;
  if (i < n) out[i] = (_Float16)in[i];
}

// vh[B,S,D] f16  ->  vT[B,D,S] f16 (B operand of ctx GEMM)
__global__ void transpose_sd_kernel(const _Float16* __restrict__ in,
                                    _Float16* __restrict__ out, long total)
{
  const long i = (long)blockIdx.x * blockDim.x + threadIdx.x;
  if (i >= total) return;
  const int  s   = (int)(i % S_);
  const long rem = i / S_;
  const int  d   = (int)(rem % D_);
  const long b   = rem / D_;
  out[i] = in[(b * S_ + s) * D_ + d];
}

// DeBERTa-v2 log-bucket relative position
__device__ __forceinline__ int rel_bucket(int i, int j) {
  const int rel = i - j;
  const int a = rel < 0 ? -rel : rel;
  if (a <= 16) return rel;  // mid = BUCKETS/2 = 16
  const float lp = ceilf(__logf((float)a * (1.f / 16.f)) *
                         (15.f / __logf(127.f / 16.f))) + 16.f;
  return rel > 0 ? (int)lp : -(int)lp;
}

// Fused: scores + scale*(c2p gather + p2c gather) -> softmax -> f16 probs.
// One block per (query row i, head h); operates on one batch's score buffer.
__global__ __launch_bounds__(256) void softmax_bias_kernel(
    const float* __restrict__ scores, const float* __restrict__ c2p,
    const float* __restrict__ p2c, _Float16* __restrict__ probs,
    int b, float scale)
{
  __shared__ float red[8];
  __shared__ float c2ps[2 * SPAN_];
  const int i = blockIdx.x, h = blockIdx.y, tid = threadIdx.x;
  const float* srow = scores + ((long)h * S_ + i) * S_;
  const float* c2pr = c2p + (((long)b * H_ + h) * S_ + i) * (2 * SPAN_);
  const float* p2cb = p2c + ((long)b * H_ + h) * S_ * (long)(2 * SPAN_);
  if (tid < 2 * SPAN_) c2ps[tid] = c2pr[tid];
  __syncthreads();

  float vals[4];
  float mx = -1e30f;
#pragma unroll
  for (int t = 0; t < 4; ++t) {
    const int j   = tid + t * 256;
    const int bk  = rel_bucket(i, j);
    int i1 = bk + SPAN_;           i1 = i1 < 0 ? 0 : (i1 > 63 ? 63 : i1);
    const int bk2 = rel_bucket(j, i);
    int i2 = -bk2 + SPAN_;         i2 = i2 < 0 ? 0 : (i2 > 63 ? 63 : i2);
    const float v = srow[j] + (c2ps[i1] + p2cb[(long)j * (2 * SPAN_) + i2]) * scale;
    vals[t] = v;
    mx = fmaxf(mx, v);
  }
  for (int o = 16; o > 0; o >>= 1) mx = fmaxf(mx, __shfl_down(mx, o));
  if ((tid & 31) == 0) red[tid >> 5] = mx;
  __syncthreads();
  if (tid == 0) { float t = red[0]; for (int k = 1; k < 8; ++k) t = fmaxf(t, red[k]); red[0] = t; }
  __syncthreads();
  mx = red[0];

  float sum = 0.f;
#pragma unroll
  for (int t = 0; t < 4; ++t) { vals[t] = __expf(vals[t] - mx); sum += vals[t]; }
  __syncthreads();
  for (int o = 16; o > 0; o >>= 1) sum += __shfl_down(sum, o);
  if ((tid & 31) == 0) red[tid >> 5] = sum;
  __syncthreads();
  if (tid == 0) { float t = 0; for (int k = 0; k < 8; ++k) t += red[k]; red[0] = t; }
  __syncthreads();
  const float inv = 1.f / red[0];

  _Float16* prow = probs + ((long)h * S_ + i) * S_;
#pragma unroll
  for (int t = 0; t < 4; ++t) prow[tid + t * 256] = (_Float16)(vals[t] * inv);
}

// ---------------------------------------------------------------------------
// Host-side dispatch
// ---------------------------------------------------------------------------
static void gemm(hipStream_t st,
                 const _Float16* A, int lda, long sAb, long sAh,
                 const _Float16* Bt, int ldb, long sBb, long sBh,
                 void* C, int ldc, long sCb, long sCh,
                 const float* Res, int ldr, long sRb, long sRh,
                 const float* bias, int M, int N, int K, int nb, int nh,
                 float scale, bool silu, bool outf16)
{
  const dim3 grid((N + BN - 1) / BN, (M + BM - 1) / BM, nb * nh);
  const bool bi = bias != nullptr, re = Res != nullptr;
#define GL(BI, RE, SI, OF) \
  gemm_wmma_kernel<BI, RE, SI, OF><<<grid, 256, 0, st>>>( \
      A, lda, sAb, sAh, Bt, ldb, sBb, sBh, C, ldc, sCb, sCh, \
      Res, ldr, sRb, sRh, bias, M, N, K, nh, scale)
  if      (bi &&  re && !silu && !outf16) GL(true,  true,  false, false);
  else if (bi && !re && !silu &&  outf16) GL(true,  false, false, true);
  else if (bi && !re &&  silu &&  outf16) GL(true,  false, true,  true);
  else if (!bi && !re && !silu && outf16) GL(false, false, false, true);
  else                                    GL(false, false, false, false);
#undef GL
}

extern "C" void kernel_launch(void* const* d_in, const int* in_sizes, int n_in,
                              void* d_out, int out_size, void* d_ws, size_t ws_size,
                              hipStream_t stream)
{
  (void)in_sizes; (void)n_in; (void)out_size; (void)ws_size;

  const float* x     = (const float*)d_in[0];
  const float* ln1g  = (const float*)d_in[1];
  const float* ln1b  = (const float*)d_in[2];
  const float* dw_w  = (const float*)d_in[3];
  const float* dw_b  = (const float*)d_in[4];
  const float* pw_w  = (const float*)d_in[5];
  const float* pw_b  = (const float*)d_in[6];
  const float* ln2g  = (const float*)d_in[7];
  const float* ln2b  = (const float*)d_in[8];
  const float* q_w   = (const float*)d_in[9];
  const float* q_b   = (const float*)d_in[10];
  const float* k_w   = (const float*)d_in[11];
  const float* k_b   = (const float*)d_in[12];
  const float* v_w   = (const float*)d_in[13];
  const float* v_b   = (const float*)d_in[14];
  const float* o_w   = (const float*)d_in[15];
  const float* o_b   = (const float*)d_in[16];
  const float* rel_e = (const float*)d_in[17];
  const float* ln3g  = (const float*)d_in[18];
  const float* ln3b  = (const float*)d_in[19];
  const float* w1    = (const float*)d_in[20];
  const float* b1    = (const float*)d_in[21];
  const float* w2    = (const float*)d_in[22];
  const float* b2    = (const float*)d_in[23];

  const long BS  = (long)B_ * S_;
  const long BSD = BS * D_;

  char* wp = (char*)d_ws;
  auto carve = [&](size_t bytes) -> void* {
    void* r = (void*)wp;
    wp += (bytes + 255) & ~(size_t)255;
    return r;
  };

  _Float16* xn1h   = (_Float16*)carve(BSD * 2);
  _Float16* a1h    = (_Float16*)carve(BSD * 2);
  float*    x1     = (float*)   carve(BSD * 4);
  _Float16* xn2h   = (_Float16*)carve(BSD * 2);
  _Float16* qh     = (_Float16*)carve(BSD * 2);
  _Float16* kh     = (_Float16*)carve(BSD * 2);
  _Float16* vh     = (_Float16*)carve(BSD * 2);
  _Float16* vTh    = (_Float16*)carve(BSD * 2);
  _Float16* relh   = (_Float16*)carve((long)2 * SPAN_ * D_ * 2);
  _Float16* poskh  = (_Float16*)carve((long)2 * SPAN_ * D_ * 2);
  _Float16* posqh  = (_Float16*)carve((long)2 * SPAN_ * D_ * 2);
  _Float16* pwWh   = (_Float16*)carve((long)D_ * D_ * 2);
  _Float16* qWh    = (_Float16*)carve((long)D_ * D_ * 2);
  _Float16* kWh    = (_Float16*)carve((long)D_ * D_ * 2);
  _Float16* vWh    = (_Float16*)carve((long)D_ * D_ * 2);
  _Float16* oWh    = (_Float16*)carve((long)D_ * D_ * 2);
  _Float16* w1h    = (_Float16*)carve((long)FF_ * D_ * 2);
  _Float16* w2h    = (_Float16*)carve((long)D_ * FF_ * 2);
  float*    c2p    = (float*)   carve((long)B_ * H_ * S_ * 64 * 4);
  float*    p2c    = (float*)   carve((long)B_ * H_ * S_ * 64 * 4);
  float*    scores = (float*)   carve((long)H_ * S_ * S_ * 4);   // per-batch reuse
  _Float16* probsh = (_Float16*)carve((long)H_ * S_ * S_ * 2);   // per-batch reuse
  _Float16* ctxh   = (_Float16*)carve(BSD * 2);
  float*    x2     = (float*)   carve(BSD * 4);
  _Float16* xn3h   = (_Float16*)carve(BSD * 2);
  _Float16* h1h    = (_Float16*)carve(BS * FF_ * 2);

  auto cvt = [&](const float* src, _Float16* dst, long n) {
    f32_to_f16_kernel<<<(int)((n + 255) / 256), 256, 0, stream>>>(src, dst, n);
  };
  cvt(pw_w, pwWh, (long)D_ * D_);
  cvt(q_w,  qWh,  (long)D_ * D_);
  cvt(k_w,  kWh,  (long)D_ * D_);
  cvt(v_w,  vWh,  (long)D_ * D_);
  cvt(o_w,  oWh,  (long)D_ * D_);
  cvt(w1,   w1h,  (long)FF_ * D_);
  cvt(w2,   w2h,  (long)D_ * FF_);
  cvt(rel_e, relh, (long)2 * SPAN_ * D_);

  const float scale = 1.f / sqrtf((float)DH_ * 3.f);

  // ---- conv block ----
  ln_f16_kernel<<<(int)BS, 256, 0, stream>>>(x, ln1g, ln1b, xn1h);
  dwconv_silu_kernel<<<(int)((BSD + 255) / 256), 256, 0, stream>>>(
      xn1h, dw_w, dw_b, a1h, BSD);
  gemm(stream, a1h, D_, 0, 0, pwWh, D_, 0, 0, x1, D_, 0, 0,
       x, D_, 0, 0, pw_b, (int)BS, D_, D_, 1, 1, 1.f, false, false);

  // ---- attention block ----
  ln_f16_kernel<<<(int)BS, 256, 0, stream>>>(x1, ln2g, ln2b, xn2h);
  gemm(stream, xn2h, D_, 0, 0, qWh, D_, 0, 0, qh, D_, 0, 0,
       nullptr, 0, 0, 0, q_b, (int)BS, D_, D_, 1, 1, 1.f, false, true);
  gemm(stream, xn2h, D_, 0, 0, kWh, D_, 0, 0, kh, D_, 0, 0,
       nullptr, 0, 0, 0, k_b, (int)BS, D_, D_, 1, 1, 1.f, false, true);
  gemm(stream, xn2h, D_, 0, 0, vWh, D_, 0, 0, vh, D_, 0, 0,
       nullptr, 0, 0, 0, v_b, (int)BS, D_, D_, 1, 1, 1.f, false, true);
  transpose_sd_kernel<<<(int)((BSD + 255) / 256), 256, 0, stream>>>(vh, vTh, BSD);

  // pos_k = rel_emb @ k_w.T + k_b ; pos_q likewise (64 x D, f16 out)
  gemm(stream, relh, D_, 0, 0, kWh, D_, 0, 0, poskh, D_, 0, 0,
       nullptr, 0, 0, 0, k_b, 2 * SPAN_, D_, D_, 1, 1, 1.f, false, true);
  gemm(stream, relh, D_, 0, 0, qWh, D_, 0, 0, posqh, D_, 0, 0,
       nullptr, 0, 0, 0, q_b, 2 * SPAN_, D_, D_, 1, 1, 1.f, false, true);

  // c2p[b,h,i,j] = q . pos_k ; p2c[b,h,i,j] = k . pos_q  (raw, scale in softmax)
  gemm(stream, qh, D_, (long)S_ * D_, DH_, poskh, D_, 0, DH_,
       c2p, 2 * SPAN_, (long)H_ * S_ * 2 * SPAN_, (long)S_ * 2 * SPAN_,
       nullptr, 0, 0, 0, nullptr, S_, 2 * SPAN_, DH_, B_, H_, 1.f, false, false);
  gemm(stream, kh, D_, (long)S_ * D_, DH_, posqh, D_, 0, DH_,
       p2c, 2 * SPAN_, (long)H_ * S_ * 2 * SPAN_, (long)S_ * 2 * SPAN_,
       nullptr, 0, 0, 0, nullptr, S_, 2 * SPAN_, DH_, B_, H_, 1.f, false, false);

  // per-batch: scores -> fused bias+softmax -> ctx
  for (int b = 0; b < B_; ++b) {
    const _Float16* qb = qh + (long)b * S_ * D_;
    const _Float16* kb = kh + (long)b * S_ * D_;
    gemm(stream, qb, D_, 0, DH_, kb, D_, 0, DH_,
         scores, S_, 0, (long)S_ * S_,
         nullptr, 0, 0, 0, nullptr, S_, S_, DH_, 1, H_, scale, false, false);
    softmax_bias_kernel<<<dim3(S_, H_), 256, 0, stream>>>(
        scores, c2p, p2c, probsh, b, scale);
    gemm(stream, probsh, S_, 0, (long)S_ * S_,
         vTh + (long)b * D_ * S_, S_, 0, (long)DH_ * S_,
         ctxh + (long)b * S_ * D_, D_, 0, DH_,
         nullptr, 0, 0, 0, nullptr, S_, DH_, S_, 1, H_, 1.f, false, true);
  }

  // o-projection + residual
  gemm(stream, ctxh, D_, 0, 0, oWh, D_, 0, 0, x2, D_, 0, 0,
       x1, D_, 0, 0, o_b, (int)BS, D_, D_, 1, 1, 1.f, false, false);

  // ---- FFN block ----
  ln_f16_kernel<<<(int)BS, 256, 0, stream>>>(x2, ln3g, ln3b, xn3h);
  gemm(stream, xn3h, D_, 0, 0, w1h, D_, 0, 0, h1h, FF_, 0, 0,
       nullptr, 0, 0, 0, b1, (int)BS, FF_, D_, 1, 1, 1.f, true, true);
  gemm(stream, h1h, FF_, 0, 0, w2h, FF_, 0, 0, (float*)d_out, D_, 0, 0,
       x2, D_, 0, 0, b2, (int)BS, D_, FF_, 1, 1, 1.f, false, false);
}
